// Block_6468220748407
// MI455X (gfx1250) — matrix operand order repaired and checked by reading, not compiled, not run on hardware
//
#include <hip/hip_runtime.h>

// ---------------------------------------------------------------------------
// Transformer block (pre-norm attn + MLP) for MI455X / gfx1250.
// GEMMs: v_wmma_f32_16x16x32_f16 (wave32), double-buffered LDS tiles fed by
// the Tensor Data Mover (A tiles, TENSORcnt) and async global->LDS loads
// (W tiles, ASYNCcnt).  f32 accumulate / residual / softmax / LN statistics.
// ---------------------------------------------------------------------------

typedef _Float16 h16;
typedef __attribute__((ext_vector_type(16))) _Float16 v16h;
typedef __attribute__((ext_vector_type(8)))  _Float16 v8h;
typedef __attribute__((ext_vector_type(8)))  float    v8f;
typedef unsigned int u32x4 __attribute__((ext_vector_type(4)));
typedef int          i32x4 __attribute__((ext_vector_type(4)));
typedef int          i32x8 __attribute__((ext_vector_type(8)));

union F16Frag { v16h v; v8h h[2]; };

static constexpr int Bsz = 8;
static constexpr int Nsq = 1024;
static constexpr int C   = 768;
static constexpr int Hh  = 12;
static constexpr int DH  = 64;
static constexpr int HID = 3072;
static constexpr int C3  = 2304;
static constexpr int Mrows = Bsz * Nsq;          // 8192

// ---- CDNA5 data-mover helpers ---------------------------------------------
__device__ __forceinline__ unsigned lds_off(const void* p) {
  // generic pointers to LDS carry the group-segment offset in the low 32 bits
  return (unsigned)(unsigned long long)p;
}

__device__ __forceinline__ void async_b128(unsigned ldsoff, const void* g) {
  asm volatile("global_load_async_to_lds_b128 %0, %1, off"
               :: "v"(ldsoff), "v"(g) : "memory");
}

__device__ __forceinline__ void wait_async0() {
  asm volatile("s_wait_asynccnt 0x0" ::: "memory");
}

// TDM: load a [128 rows x 64B] tile (row stride = lda halves) into LDS.
__device__ __forceinline__ void tdm_load_tile_A(const h16* gsrc, unsigned ldsoff,
                                                unsigned stride_units /*8B units*/) {
  const unsigned long long ga = (unsigned long long)gsrc;
  u32x4 g0;
  g0[0] = 1u;                                                // count=1 (user D#)
  g0[1] = ldsoff;                                            // lds_addr (bytes)
  g0[2] = (unsigned)ga;                                      // global_addr[31:0]
  g0[3] = (unsigned)((ga >> 32) & 0x01FFFFFFu) | (2u << 30); // addr[56:32] | type=2
  i32x8 g1;
  g1[0] = (int)(3u << 16);            // workgroup_mask=0, data_size=8B
  g1[1] = (int)0xFFFF0000u;           // tensor_dim0[15:0]  (huge -> no OOB clip)
  g1[2] = (int)0xFFFF7FFFu;           // tensor_dim0[31:16] / tensor_dim1[15:0]
  g1[3] = (int)((8u << 16) | 0x7FFFu);// tile_dim0=8 (64B row) / tensor_dim1[31:16]
  g1[4] = 128;                        // tile_dim1=128 rows (tile_dim2=0 unused)
  g1[5] = (int)stride_units;          // tensor_dim0_stride[31:0]
  g1[6] = 0;                          // stride hi / dim1_stride lo
  g1[7] = 0;
  const i32x4 z4 = {0, 0, 0, 0};      // groups 2/3 unused (2D tile)
  const i32x8 z8 = {0, 0, 0, 0, 0, 0, 0, 0};
  __builtin_amdgcn_tensor_load_to_lds(g0, g1, z4, z4, z8, 0);
}

// ---------------------------------------------------------------------------
// f32 -> f16 conversion (weights)
// ---------------------------------------------------------------------------
__global__ void cvt_f32_f16_kernel(const float* __restrict__ src,
                                   h16* __restrict__ dst, int n) {
  int i = blockIdx.x * 256 + threadIdx.x;
  if (i < n) dst[i] = (h16)src[i];
}

// ---------------------------------------------------------------------------
// LayerNorm (per-row, C=768), f32 in -> f16 out
// ---------------------------------------------------------------------------
__global__ void layernorm_f16_kernel(const float* __restrict__ x,
                                     const float* __restrict__ g,
                                     const float* __restrict__ bta,
                                     h16* __restrict__ out) {
  __shared__ float rs[256];
  __shared__ float rq[256];
  const int row = blockIdx.x;
  const int tid = threadIdx.x;
  const float* xr = x + (size_t)row * C;
  float v[3], s = 0.f, q = 0.f;
#pragma unroll
  for (int i = 0; i < 3; ++i) {
    v[i] = xr[tid + i * 256];
    s += v[i];
    q += v[i] * v[i];
  }
  rs[tid] = s; rq[tid] = q;
  __syncthreads();
  for (int o = 128; o > 0; o >>= 1) {
    if (tid < o) { rs[tid] += rs[tid + o]; rq[tid] += rq[tid + o]; }
    __syncthreads();
  }
  const float mu  = rs[0] * (1.0f / C);
  const float var = rq[0] * (1.0f / C) - mu * mu;
  const float r   = rsqrtf(var + 1e-5f);
  h16* orow = out + (size_t)row * C;
#pragma unroll
  for (int i = 0; i < 3; ++i) {
    const int c = tid + i * 256;
    orow[c] = (h16)((v[i] - mu) * r * g[c] + bta[c]);
  }
}

// ---------------------------------------------------------------------------
// Tiled WMMA GEMM:  C[M,N] = A[M,K](f16) * W[K,N](f16)  + epilogue
//   mode 0: out16 = result              (qkv: no bias)
//   mode 1: out32 = resid + result+bias (proj / fc2)
//   mode 2: out16 = gelu(result+bias)   (fc1)
// 256 threads = 8 waves; BM=BN=128, BK=32; wave tile 64x32 (4x2 WMMA tiles).
// A tiles via TDM (tensor_load_to_lds), W tiles via async LDS loads, both
// double-buffered against the WMMA consumer.
// ---------------------------------------------------------------------------
__global__ void gemm_f16_kernel(const h16* __restrict__ A, int lda,
                                const h16* __restrict__ W, int ldw,
                                int K,
                                const float* __restrict__ bias,
                                const float* __restrict__ resid,
                                float* __restrict__ out32,
                                h16* __restrict__ out16,
                                int ldo, int mode) {
  extern __shared__ char smem[];
  h16* As = (h16*)smem;                  // 2 x [128][32]  (m, k) row-major
  h16* Wt = (h16*)(smem + 2 * 8192);     // 2 x [128][32]  (n, k) n-major
  h16* Ws = (h16*)(smem + 4 * 8192);     // [32][128] raw staging for W tile

  const int tid  = threadIdx.x;
  const int lane = tid & 31;
  const int wid  = tid >> 5;
  const int l16  = lane >> 4;
  const int lm   = lane & 15;

  const int m0 = blockIdx.y * 128;
  const int n0 = blockIdx.x * 128;
  const int waveM = (wid >> 2) * 64;
  const int waveN = (wid & 3) * 32;

  const unsigned strideA = (unsigned)lda >> 2;   // row stride in 8B units

  // stage W tile kb2 (raw, row-major) into Ws via async LDS loads
  auto stageW = [&](int kb2) {
    const int k0 = kb2 * 32;
    for (int c = tid; c < 512; c += 256) {
      const int kr = c >> 4;           // 0..31
      const int nc = (c & 15) * 8;     // 0..120
      async_b128(lds_off(&Ws[kr * 128 + nc]),
                 (const void*)&W[(size_t)(k0 + kr) * ldw + n0 + nc]);
    }
  };
  // transpose own staged chunks Ws -> dst[n][k] (only needs own asynccnt)
  auto xposeW = [&](h16* dst) {
    for (int c = tid; c < 512; c += 256) {
      const int kr = c >> 4;
      const int nc = (c & 15) * 8;
      v8h w = *(const v8h*)&Ws[kr * 128 + nc];
#pragma unroll
      for (int e = 0; e < 8; ++e) dst[(nc + e) * 32 + kr] = w[e];
    }
  };

  v8f acc[4][2] = {};
  const int nk = K / 32;

  // ---- prologue: fill buffer 0 ----
  if (wid == 0)
    tdm_load_tile_A(&A[(size_t)m0 * lda], lds_off(As), strideA);
  stageW(0);
  wait_async0();
  xposeW(Wt);
  if (wid == 0) __builtin_amdgcn_s_wait_tensorcnt(0);
  __syncthreads();

  for (int kb = 0; kb < nk; ++kb) {
    const int cur = kb & 1;
    const h16* Asc = As + cur * 4096;
    const h16* Wtc = Wt + cur * 4096;

    // kick off fills for tile kb+1 (TDM + async overlap the WMMAs below)
    if (kb + 1 < nk) {
      const int nxt = cur ^ 1;
      if (wid == 0)
        tdm_load_tile_A(&A[(size_t)m0 * lda + (kb + 1) * 32],
                        lds_off(As + nxt * 4096), strideA);
      stageW(kb + 1);
    }

    // consume tile kb
    F16Frag a[4], b[2];
#pragma unroll
    for (int mt = 0; mt < 4; ++mt) {
      const h16* base = &Asc[(waveM + mt * 16 + lm) * 32 + (l16 ? 8 : 0)];
      a[mt].h[0] = *(const v8h*)base;
      a[mt].h[1] = *(const v8h*)(base + 16);
    }
#pragma unroll
    for (int nt = 0; nt < 2; ++nt) {
      const h16* base = &Wtc[(waveN + nt * 16 + lm) * 32 + (l16 ? 8 : 0)];
      b[nt].h[0] = *(const v8h*)base;
      b[nt].h[1] = *(const v8h*)(base + 16);
    }
#pragma unroll
    for (int mt = 0; mt < 4; ++mt)
#pragma unroll
      for (int nt = 0; nt < 2; ++nt)
        acc[mt][nt] = __builtin_amdgcn_wmma_f32_16x16x32_f16(
            false, a[mt].v, false, b[nt].v, (short)0, acc[mt][nt], false, false);

    // finish the next tile's fills
    if (kb + 1 < nk) {
      wait_async0();
      xposeW(Wt + (cur ^ 1) * 4096);
      if (wid == 0) __builtin_amdgcn_s_wait_tensorcnt(0);
    }
    __syncthreads();
  }

  // ---- epilogue: VGPR r -> row (r + 8*l16), lanes 0..15 -> cols ----
#pragma unroll
  for (int mt = 0; mt < 4; ++mt) {
#pragma unroll
    for (int nt = 0; nt < 2; ++nt) {
      const int n = n0 + waveN + nt * 16 + lm;
      const float bv = bias ? bias[n] : 0.0f;
#pragma unroll
      for (int r = 0; r < 8; ++r) {
        const int m = m0 + waveM + mt * 16 + r + l16 * 8;
        const float vv = acc[mt][nt][r] + bv;
        const size_t o = (size_t)m * ldo + n;
        if (mode == 0) {
          out16[o] = (h16)vv;
        } else if (mode == 1) {
          out32[o] = resid[o] + vv;
        } else {
          const float gv = 0.5f * vv * (1.0f + erff(vv * 0.70710678118f));
          out16[o] = (h16)gv;
        }
      }
    }
  }
}

// ---------------------------------------------------------------------------
// Masked attention: one workgroup per (b, h, 16-query block).
// 256 threads = 8 waves; each wave owns a 128-key slice.
// ---------------------------------------------------------------------------
__global__ void attn_kernel(const h16* __restrict__ qkv,
                            const int* __restrict__ mask,
                            h16* __restrict__ o16) {
  extern __shared__ char smem[];
  float* S  = (float*)smem;                         // 16*1024 f32  (64 KB)
  h16*   P  = (h16*)(smem + 65536);                 // 16*1024 f16  (32 KB)
  h16*   Vt = (h16*)(smem + 65536 + 32768);         // 64*1024 f16  (128 KB)
  float* red     = (float*)(smem + 65536 + 32768 + 131072); // 16*16 f32
  float* rowstat = red + 256;                               // 16 f32

  const int tid  = threadIdx.x;
  const int lane = tid & 31;
  const int wid  = tid >> 5;
  const int l16  = lane >> 4;
  const int lm   = lane & 15;

  const int bx = blockIdx.x;
  const int qb = bx & 63;
  const int h  = (bx >> 6) % Hh;
  const int b  = bx / (64 * Hh);

  const size_t rs    = C3;
  const size_t baseQ = (size_t)b * Nsq * rs + (size_t)h * DH;
  const size_t baseK = baseQ + C;
  const size_t baseV = baseQ + 2 * C;

  // ---- preload V transposed: Vt[d][key] ----
  for (int c = tid; c < 8192; c += 256) {
    const int key = c >> 3;
    const int d0  = (c & 7) * 8;
    v8h v = *(const v8h*)&qkv[baseV + (size_t)key * rs + d0];
#pragma unroll
    for (int e = 0; e < 8; ++e) Vt[(d0 + e) * 1024 + key] = v[e];
  }

  // ---- Q fragments straight from global ----
  F16Frag aq[2];
  {
    const h16* qrow = &qkv[baseQ + (size_t)(qb * 16 + lm) * rs];
#pragma unroll
    for (int s = 0; s < 2; ++s) {
      const int c0 = s * 32 + (l16 ? 8 : 0);
      aq[s].h[0] = *(const v8h*)(qrow + c0);
      aq[s].h[1] = *(const v8h*)(qrow + c0 + 16);
    }
  }

  // ---- S = (Q K^T) * scale, key mask -> -inf ----
  const float scale = 0.125f;
#pragma unroll
  for (int kt = 0; kt < 8; ++kt) {
    const int key = wid * 128 + kt * 16 + lm;
    const h16* krow = &qkv[baseK + (size_t)key * rs];
    F16Frag bk[2];
#pragma unroll
    for (int s = 0; s < 2; ++s) {
      const int c0 = s * 32 + (l16 ? 8 : 0);
      bk[s].h[0] = *(const v8h*)(krow + c0);
      bk[s].h[1] = *(const v8h*)(krow + c0 + 16);
    }
    v8f acc = {};
    acc = __builtin_amdgcn_wmma_f32_16x16x32_f16(false, aq[0].v, false, bk[0].v,
                                                 (short)0, acc, false, false);
    acc = __builtin_amdgcn_wmma_f32_16x16x32_f16(false, aq[1].v, false, bk[1].v,
                                                 (short)0, acc, false, false);
    const int mv = mask[b * Nsq + key];
#pragma unroll
    for (int r = 0; r < 8; ++r) {
      const int row = r + l16 * 8;
      S[row * 1024 + key] = mv ? -__builtin_inff() : acc[r] * scale;
    }
  }
  __syncthreads();

  // ---- masked softmax per row ----
  {
    const int row = tid >> 4;
    const int c   = tid & 15;
    float m = -__builtin_inff();
    for (int j = 0; j < 64; ++j) m = fmaxf(m, S[row * 1024 + c + j * 16]);
    red[row * 16 + c] = m;
    __syncthreads();
    if (c == 0) {
      float mm = red[row * 16];
      for (int j = 1; j < 16; ++j) mm = fmaxf(mm, red[row * 16 + j]);
      rowstat[row] = mm;
    }
    __syncthreads();
    const float rmax = rowstat[row];
    float s = 0.0f;
    for (int j = 0; j < 64; ++j) {
      const int i = row * 1024 + c + j * 16;
      const float e = __expf(S[i] - rmax);
      S[i] = e;
      s += e;
    }
    red[row * 16 + c] = s;
    __syncthreads();
    if (c == 0) {
      float ss = 0.0f;
      for (int j = 0; j < 16; ++j) ss += red[row * 16 + j];
      rowstat[row] = ss;
    }
    __syncthreads();
    float inv = rowstat[row];
    inv = inv > 0.0f ? 1.0f / inv : 0.0f;
    for (int j = 0; j < 64; ++j) {
      const int i = row * 1024 + c + j * 16;
      P[i] = (h16)(S[i] * inv);
    }
  }
  __syncthreads();

  // ---- O_partial = P[:, wave slice] x V slice ----
  v8f accO[4] = {};
#pragma unroll
  for (int ks = 0; ks < 4; ++ks) {
    const int kbase = wid * 128 + ks * 32;
    F16Frag ap;
    const h16* prow = &P[lm * 1024 + kbase + (l16 ? 8 : 0)];
    ap.h[0] = *(const v8h*)prow;
    ap.h[1] = *(const v8h*)(prow + 16);
#pragma unroll
    for (int nt = 0; nt < 4; ++nt) {
      F16Frag bv;
      const h16* vrow = &Vt[(nt * 16 + lm) * 1024 + kbase + (l16 ? 8 : 0)];
      bv.h[0] = *(const v8h*)vrow;
      bv.h[1] = *(const v8h*)(vrow + 16);
      accO[nt] = __builtin_amdgcn_wmma_f32_16x16x32_f16(
          false, ap.v, false, bv.v, (short)0, accO[nt], false, false);
    }
  }
  __syncthreads();
  float* Op = S;   // reuse S region: [wave][16][64]
#pragma unroll
  for (int nt = 0; nt < 4; ++nt)
#pragma unroll
    for (int r = 0; r < 8; ++r) {
      const int row = r + l16 * 8;
      const int d   = nt * 16 + lm;
      Op[(wid * 16 + row) * 64 + d] = accO[nt][r];
    }
  __syncthreads();
  for (int i = tid; i < 1024; i += 256) {
    const int row = i >> 6;
    const int d   = i & 63;
    float s = 0.0f;
#pragma unroll
    for (int w = 0; w < 8; ++w) s += Op[(w * 16 + row) * 64 + d];
    o16[((size_t)b * Nsq + qb * 16 + row) * C + h * DH + d] = (h16)s;
  }
}

// ---------------------------------------------------------------------------
// Host-side orchestration
// ---------------------------------------------------------------------------
extern "C" void kernel_launch(void* const* d_in, const int* in_sizes, int n_in,
                              void* d_out, int out_size, void* d_ws, size_t ws_size,
                              hipStream_t stream) {
  const float* x      = (const float*)d_in[0];
  const int*   mask   = (const int*)d_in[1];
  const float* ln1_g  = (const float*)d_in[2];
  const float* ln1_b  = (const float*)d_in[3];
  const float* qkv_w  = (const float*)d_in[4];
  const float* proj_w = (const float*)d_in[5];
  const float* proj_b = (const float*)d_in[6];
  const float* ln2_g  = (const float*)d_in[7];
  const float* ln2_b  = (const float*)d_in[8];
  const float* fc1_w  = (const float*)d_in[9];
  const float* fc1_b  = (const float*)d_in[10];
  const float* fc2_w  = (const float*)d_in[11];
  const float* fc2_b  = (const float*)d_in[12];
  float* out = (float*)d_out;

  char* ws = (char*)d_ws;
  h16*   Wqkv  = (h16*)(ws + 0);                       // 768*2304 f16
  h16*   Wproj = (h16*)(ws + 3538944);                 // 768*768  f16
  h16*   Wfc1  = (h16*)(ws + 4718592);                 // 768*3072 f16
  h16*   Wfc2  = (h16*)(ws + 9437184);                 // 3072*768 f16
  h16*   A16   = (h16*)(ws + 14155776);                // 8192*3072 f16
  h16*   QKV16 = (h16*)(ws + 64487424);                // 8192*2304 f16
  h16*   O16   = (h16*)(ws + 102236160);               // 8192*768 f16
  float* X1    = (float*)(ws + 114819072);             // 8192*768 f32

  cvt_f32_f16_kernel<<<(C * C3 + 255) / 256, 256, 0, stream>>>(qkv_w, Wqkv, C * C3);
  cvt_f32_f16_kernel<<<(C * C + 255) / 256, 256, 0, stream>>>(proj_w, Wproj, C * C);
  cvt_f32_f16_kernel<<<(C * HID + 255) / 256, 256, 0, stream>>>(fc1_w, Wfc1, C * HID);
  cvt_f32_f16_kernel<<<(HID * C + 255) / 256, 256, 0, stream>>>(fc2_w, Wfc2, HID * C);

  layernorm_f16_kernel<<<Mrows, 256, 0, stream>>>(x, ln1_g, ln1_b, A16);

  const size_t gemm_lds = (2 * 4096 + 2 * 4096 + 4096) * sizeof(h16);  // 40 KB
  gemm_f16_kernel<<<dim3(C3 / 128, Mrows / 128), 256, gemm_lds, stream>>>(
      A16, C, Wqkv, C3, C, nullptr, nullptr, nullptr, QKV16, C3, 0);

  const size_t attn_lds = 65536 + 32768 + 131072 + (256 + 16) * sizeof(float);
  attn_kernel<<<Bsz * Hh * (Nsq / 16), 256, attn_lds, stream>>>(QKV16, mask, O16);

  gemm_f16_kernel<<<dim3(C / 128, Mrows / 128), 256, gemm_lds, stream>>>(
      O16, C, Wproj, C, C, proj_b, x, X1, nullptr, C, 1);

  layernorm_f16_kernel<<<Mrows, 256, 0, stream>>>(X1, ln2_g, ln2_b, O16);

  gemm_f16_kernel<<<dim3(HID / 128, Mrows / 128), 256, gemm_lds, stream>>>(
      O16, C, Wfc1, HID, C, fc1_b, nullptr, nullptr, A16, HID, 2);

  gemm_f16_kernel<<<dim3(C / 128, Mrows / 128), 256, gemm_lds, stream>>>(
      A16, HID, Wfc2, C, HID, fc2_b, X1, out, nullptr, C, 1);
}
